// HyperLinear_50208167690693
// MI455X (gfx1250) — compile-verified, matrix operand order checked by reading, not run on hardware
//
#include <hip/hip_runtime.h>

// HyperLinear for MI455X (gfx1250, wave32, WMMA).
// Algebraic restructure avoids the (B,256,256) per-sample weight tensor:
//   y = x@target_w^T + t@fcwe_w^T + fcwe_b*rowsum(x) + bias
//   t = (h-contraction of G) + x@C ;  G = x@Wp  (Wp = fcwp_w rearranged)
// All GEMMs run on v_wmma_f32_16x16x32_bf16 (bf16 in, f32 accumulate).
// Each wave computes a 32x64 output block (2 A frags x 4 B frags = 8
// accumulator chains): 8 wmma per 12 b128 loads, and enough independent
// chains for intra-wave load/wmma overlap (grid is only ~1 wave/SIMD, so
// cross-wave occupancy cannot hide latency). __launch_bounds__(256,1)
// relaxes the VGPR cap so all fragments stay live (no single-buffer
// s_wait_loadcnt 0x0 serialization as seen in round 2).

#define B_    2048
#define NP_   8
#define NL_   4
#define CIN_  256
#define COUT_ 256
#define MID_  64

typedef __attribute__((ext_vector_type(16))) __bf16        v16bf;
typedef __attribute__((ext_vector_type(8)))  float         v8f;
typedef __attribute__((ext_vector_type(4)))  unsigned int  v4u;

union BF16Frag { v16bf v; v4u q[2]; };

__device__ __forceinline__ unsigned short f2bf(float f) {
    unsigned int u = __float_as_uint(f);
    u += 0x7FFFu + ((u >> 16) & 1u);   // round-to-nearest-even
    return (unsigned short)(u >> 16);
}

// One wave computes a (MT*16) x (NT*16) f32 block of
// A(MxK,row-major bf16) * Bn^T, where Bn is stored [N][K] row-major bf16
// (per-lane K-runs contiguous). Fragment mapping per CDNA5 ISA 7.12.2:
//   A: lane L holds row m+(L&15); element j -> K = (j&7) + (j>>3)*16 + (L>>4)*8
//   B: lane L holds col n+(L&15); same K mapping
//   C/D: lane L, vgpr r -> (M = m0 + r + (L>>4)*8, N = n0 + (L&15))
template <int K, int MT, int NT>
__device__ __forceinline__ void wave_gemm_block(const unsigned short* __restrict__ A,
                                                const unsigned short* __restrict__ Bn,
                                                int m0, int n0, v8f acc[MT][NT]) {
    const int lane = threadIdx.x & 31;
    const int koff = (lane >> 4) * 8;
    const unsigned short* pa[MT];
    const unsigned short* pb[NT];
#pragma unroll
    for (int s = 0; s < MT; ++s)
        pa[s] = A + (size_t)(m0 + s * 16 + (lane & 15)) * K + koff;
#pragma unroll
    for (int t = 0; t < NT; ++t)
        pb[t] = Bn + (size_t)(n0 + t * 16 + (lane & 15)) * K + koff;
#pragma unroll
    for (int kk = 0; kk < K; kk += 32) {
        BF16Frag a[MT], b[NT];
#pragma unroll
        for (int s = 0; s < MT; ++s) {
            a[s].q[0] = *(const v4u*)(pa[s]);        // K: koff .. koff+7
            a[s].q[1] = *(const v4u*)(pa[s] + 16);   // K: koff+16 .. koff+23
            pa[s] += 32;
        }
#pragma unroll
        for (int t = 0; t < NT; ++t) {
            b[t].q[0] = *(const v4u*)(pb[t]);
            b[t].q[1] = *(const v4u*)(pb[t] + 16);
            pb[t] += 32;
        }
#pragma unroll
        for (int s = 0; s < MT; ++s)
#pragma unroll
            for (int t = 0; t < NT; ++t)
                acc[s][t] = __builtin_amdgcn_wmma_f32_16x16x32_bf16(
                    false, a[s].v, false, b[t].v, (short)0, acc[s][t], false, false);
    }
}

// ---------------- prep kernels ----------------

__global__ void k_f2bf(const float* __restrict__ src, unsigned short* __restrict__ dst, int n) {
    int i = blockIdx.x * blockDim.x + threadIdx.x;
    if (i < n) dst[i] = f2bf(src[i]);
}

// c[j] = fcwp_b[j] + fcwi_b[j] + sum_l index[l]*fcwi_w[j,l]; store rearranged
// as B-matrix [n=m][k=i]: c_b[m*CIN + i], where j = i*MID + m.
__global__ void k_prep_c(const float* __restrict__ fcwp_b, const float* __restrict__ fcwi_w,
                         const float* __restrict__ fcwi_b, const float* __restrict__ idxv,
                         unsigned short* __restrict__ c_b) {
    int j = blockIdx.x * blockDim.x + threadIdx.x;
    if (j >= MID_ * CIN_) return;
    float v = fcwp_b[j] + fcwi_b[j];
#pragma unroll
    for (int l = 0; l < NL_; ++l) v += fcwi_w[j * NL_ + l] * idxv[l];
    int i = j / MID_, m = j % MID_;
    c_b[m * CIN_ + i] = f2bf(v);
}

// Wp as B-matrix [n=m*8+p][k=i]: wp_b[n*CIN+i] = fcwp_w[(i*MID+m)*NP + p]
__global__ void k_prep_wp(const float* __restrict__ fcwp_w, unsigned short* __restrict__ wp_b) {
    int t = blockIdx.x * blockDim.x + threadIdx.x;
    if (t >= (MID_ * NP_) * CIN_) return;
    int n = t / CIN_, i = t % CIN_;
    int m = n / NP_, p = n % NP_;
    wp_b[(size_t)n * CIN_ + i] = f2bf(fcwp_w[((size_t)i * MID_ + m) * NP_ + p]);
}

__global__ void k_prep_bconst(const float* __restrict__ target_b, const float* __restrict__ fcbp_b,
                              const float* __restrict__ fcbi_w, const float* __restrict__ fcbi_b,
                              const float* __restrict__ idxv, float* __restrict__ bconst) {
    int o = blockIdx.x * blockDim.x + threadIdx.x;
    if (o >= COUT_) return;
    float v = target_b[o] + fcbp_b[o] + fcbi_b[o];
#pragma unroll
    for (int l = 0; l < NL_; ++l) v += fcbi_w[o * NL_ + l] * idxv[l];
    bconst[o] = v;
}

__global__ void k_sx(const float* __restrict__ x, float* __restrict__ sx) {
    int b = blockIdx.x * blockDim.x + threadIdx.x;
    if (b >= B_) return;
    float s = 0.f;
    for (int i = 0; i < CIN_; ++i) s += x[(size_t)b * CIN_ + i];
    sx[b] = s;
}

// ---------------- GEMM kernels ----------------

// C(MxN,f32) = A(Mx256 bf16) * Bn^T, Bn stored [N][256] bf16.
// One 32x64 block per wave; M multiple of 32, N multiple of 64.
__global__ void __launch_bounds__(256, 1)
k_gemm_k256(const unsigned short* __restrict__ A,
            const unsigned short* __restrict__ Bn,
            float* __restrict__ C, int N, int blocks) {
    int wave = (blockIdx.x * blockDim.x + threadIdx.x) >> 5;
    if (wave >= blocks) return;
    int nsn = N >> 6;                       // 64-wide blocks per row strip
    int tm = wave / nsn, tn = wave % nsn;
    int m0 = tm * 32, n0 = tn * 64;
    v8f acc[2][4];
#pragma unroll
    for (int s = 0; s < 2; ++s)
#pragma unroll
        for (int t = 0; t < 4; ++t) acc[s][t] = (v8f){0.f,0.f,0.f,0.f,0.f,0.f,0.f,0.f};
    wave_gemm_block<CIN_, 2, 4>(A, Bn, m0, n0, acc);
    int lane = threadIdx.x & 31;
#pragma unroll
    for (int s = 0; s < 2; ++s)
#pragma unroll
    for (int t = 0; t < 4; ++t) {
        int col = n0 + t * 16 + (lane & 15);
#pragma unroll
        for (int r = 0; r < 8; ++r) {
            int row = m0 + s * 16 + r + ((lane >> 4) << 3);
            C[(size_t)row * N + col] = acc[s][t][r];
        }
    }
}

// t[b,m] = T1[b,m] + sum_p h[b,p]*G[b, m*8+p]  -> bf16
__global__ void k_t(const float* __restrict__ h, const float* __restrict__ T1,
                    const float* __restrict__ G, unsigned short* __restrict__ t_bf) {
    int idx = blockIdx.x * blockDim.x + threadIdx.x;
    if (idx >= B_ * MID_) return;
    int b = idx >> 6, m = idx & 63;
    float v = T1[idx];
    const float* hp = h + (size_t)b * NP_;
    const float* gp = G + (size_t)b * (MID_ * NP_) + m * NP_;
#pragma unroll
    for (int p = 0; p < NP_; ++p) v += hp[p] * gp[p];
    t_bf[idx] = f2bf(v);
}

// y = x@target_w^T (K=256) + t@fcwe_w^T (K=64) + bconst + fcwe_b*sx + h@fcbp_w^T
// One 32x64 block per wave (COUT=256 -> 4 blocks across N).
__global__ void __launch_bounds__(256, 1)
k_out(const unsigned short* __restrict__ xb, const unsigned short* __restrict__ twb,
      const unsigned short* __restrict__ tb, const unsigned short* __restrict__ web,
      const float* __restrict__ bconst, const float* __restrict__ sx,
      const float* __restrict__ h, const float* __restrict__ fcbp_w,
      const float* __restrict__ fcwe_b, float* __restrict__ y) {
    int wave = (blockIdx.x * blockDim.x + threadIdx.x) >> 5;
    const int nsn = COUT_ / 64;                  // 4 blocks across N
    int tm = wave / nsn, tn = wave % nsn;        // tm < 64
    int m0 = tm * 32, n0 = tn * 64;
    v8f acc[2][4];
#pragma unroll
    for (int s = 0; s < 2; ++s)
#pragma unroll
        for (int t = 0; t < 4; ++t) acc[s][t] = (v8f){0.f,0.f,0.f,0.f,0.f,0.f,0.f,0.f};
    wave_gemm_block<CIN_, 2, 4>(xb, twb, m0, n0, acc);   // x @ target_w^T
    wave_gemm_block<MID_, 2, 4>(tb, web, m0, n0, acc);   // t @ fcwe_w^T
    int lane = threadIdx.x & 31;
#pragma unroll
    for (int t = 0; t < 4; ++t) {
        int col = n0 + t * 16 + (lane & 15);
        float bc = bconst[col];
        float web_c = fcwe_b[col];
        const float* bp = fcbp_w + (size_t)col * NP_;
#pragma unroll
        for (int s = 0; s < 2; ++s)
#pragma unroll
        for (int r = 0; r < 8; ++r) {
            int row = m0 + s * 16 + r + ((lane >> 4) << 3);
            float v = acc[s][t][r] + bc + web_c * sx[row];
            const float* hp = h + (size_t)row * NP_;
#pragma unroll
            for (int p = 0; p < NP_; ++p) v += hp[p] * bp[p];
            y[(size_t)row * COUT_ + col] = v;
        }
    }
}

// ---------------- launch ----------------

// Workspace layout (bytes, all 256B-aligned)
#define OFF_XB     0u          // x bf16:      2048*256*2 = 1048576
#define OFF_TWB    1048576u    // target_w:    256*256*2  = 131072
#define OFF_WEB    1179648u    // fcwe_w:      256*64*2   = 32768
#define OFF_WPB    1212416u    // Wp [512][256]*2         = 262144
#define OFF_CB     1474560u    // C  [64][256]*2          = 32768
#define OFF_BCONST 1507328u    // f32 256                 = 1024
#define OFF_SX     1508352u    // f32 2048                = 8192
#define OFF_G      1516544u    // f32 2048*512            = 4194304
#define OFF_T1     5710848u    // f32 2048*64             = 524288
#define OFF_TBF    6235136u    // bf16 2048*64            = 262144
                               // total ~6.2 MB

extern "C" void kernel_launch(void* const* d_in, const int* in_sizes, int n_in,
                              void* d_out, int out_size, void* d_ws, size_t ws_size,
                              hipStream_t stream) {
    const float* x        = (const float*)d_in[0];
    const float* h        = (const float*)d_in[1];
    const float* idxv     = (const float*)d_in[2];
    const float* target_w = (const float*)d_in[3];
    const float* target_b = (const float*)d_in[4];
    const float* fcwp_w   = (const float*)d_in[5];
    const float* fcwp_b   = (const float*)d_in[6];
    const float* fcwi_w   = (const float*)d_in[7];
    const float* fcwi_b   = (const float*)d_in[8];
    const float* fcwe_w   = (const float*)d_in[9];
    const float* fcwe_b   = (const float*)d_in[10];
    const float* fcbp_w   = (const float*)d_in[11];
    const float* fcbp_b   = (const float*)d_in[12];
    const float* fcbi_w   = (const float*)d_in[13];
    const float* fcbi_b   = (const float*)d_in[14];
    float* y = (float*)d_out;

    char* ws = (char*)d_ws;
    unsigned short* xb   = (unsigned short*)(ws + OFF_XB);
    unsigned short* twb  = (unsigned short*)(ws + OFF_TWB);
    unsigned short* web  = (unsigned short*)(ws + OFF_WEB);
    unsigned short* wpb  = (unsigned short*)(ws + OFF_WPB);
    unsigned short* cb   = (unsigned short*)(ws + OFF_CB);
    float*          bconst = (float*)(ws + OFF_BCONST);
    float*          sx   = (float*)(ws + OFF_SX);
    float*          G    = (float*)(ws + OFF_G);
    float*          T1   = (float*)(ws + OFF_T1);
    unsigned short* tbf  = (unsigned short*)(ws + OFF_TBF);

    const int TB = 256;

    // prep: bf16 conversions + constant folds
    k_f2bf<<<(B_ * CIN_ + TB - 1) / TB, TB, 0, stream>>>(x, xb, B_ * CIN_);
    k_f2bf<<<(COUT_ * CIN_ + TB - 1) / TB, TB, 0, stream>>>(target_w, twb, COUT_ * CIN_);
    k_f2bf<<<(COUT_ * MID_ + TB - 1) / TB, TB, 0, stream>>>(fcwe_w, web, COUT_ * MID_);
    k_prep_c<<<(MID_ * CIN_ + TB - 1) / TB, TB, 0, stream>>>(fcwp_b, fcwi_w, fcwi_b, idxv, cb);
    k_prep_wp<<<(MID_ * NP_ * CIN_ + TB - 1) / TB, TB, 0, stream>>>(fcwp_w, wpb);
    k_prep_bconst<<<1, TB, 0, stream>>>(target_b, fcbp_b, fcbi_w, fcbi_b, idxv, bconst);
    k_sx<<<(B_ + TB - 1) / TB, TB, 0, stream>>>(x, sx);

    // G = x @ Wp : M=2048, N=512, K=256 -> 64*8 = 512 wave-blocks
    {
        int blocks = (B_ / 32) * ((MID_ * NP_) / 64);
        k_gemm_k256<<<(blocks * 32) / TB, TB, 0, stream>>>(xb, wpb, G, MID_ * NP_, blocks);
    }
    // T1 = x @ C : M=2048, N=64, K=256 -> 64 wave-blocks
    {
        int blocks = (B_ / 32) * (MID_ / 64);
        k_gemm_k256<<<(blocks * 32) / TB, TB, 0, stream>>>(xb, cb, T1, MID_, blocks);
    }
    // t = T1 + h·G  (bf16)
    k_t<<<(B_ * MID_ + TB - 1) / TB, TB, 0, stream>>>(h, T1, G, tbf);

    // y = x@target_w^T + t@fcwe_w^T + epilogue : 64*4 = 256 wave-blocks
    {
        int blocks = (B_ / 32) * (COUT_ / 64);
        k_out<<<(blocks * 32) / TB, TB, 0, stream>>>(xb, twb, tbf, web, bconst, sx,
                                                     h, fcbp_w, fcwe_b, y);
    }
    (void)in_sizes; (void)n_in; (void)out_size; (void)ws_size;
}
